// ValueLayerFast_1245540516056
// MI455X (gfx1250) — matrix operand order repaired and checked by reading, not compiled, not run on hardware
//
#include <hip/hip_runtime.h>

typedef __attribute__((ext_vector_type(16))) __bf16 v16bf;
typedef __attribute__((ext_vector_type(8)))  float  v8f;

#define HEADS   8
#define NI      48      // i-dimension actually used (rest of psi is zero)
#define NJ      48      // j-dimension actually used
#define OBS_LD  64      // physical leading dims of obs_re (8,64,64)
#define PITCH   72      // LDS row pitch in bf16: 144B=36 dwords -> 16B-aligned frags,
                        // conflict-free (36*i mod 64 distinct over i=0..15)

__device__ __forceinline__ unsigned f2bf(float f) {
    unsigned u = __float_as_uint(f);
    u += 0x7FFFu + ((u >> 16) & 1u);   // round-to-nearest-even to bf16
    return u >> 16;
}
__device__ __forceinline__ unsigned pk2(float lo, float hi) {
    return f2bf(lo) | (f2bf(hi) << 16);
}

union Frag16 {                 // 16 bf16 = 8 VGPRs, per-lane WMMA A/B operand
    v16bf v;
    unsigned short s[16];
    unsigned u[8];
    uint4 q[2];
};

__global__ __launch_bounds__(256) void quadform_wmma_kernel(
    const float* __restrict__ x,        // (n_tok, 48)
    const float* __restrict__ obs_re,   // (8, 64, 64), only [.., :48, :48] used
    float* __restrict__ out,            // (n_tok, 8)
    int n_tok)
{
    // bf16 obs, layout [h][i][j], j 0..63 stored (j>=48 zeroed), pitch 72.
    __shared__ unsigned short obs_s[HEADS * NI * PITCH];    // 55296 B

    const int tid = threadIdx.x;

    // ---- Stage obs_re -> LDS bf16, vectorized: one 8-j chunk per iteration ----
    // chunks c=0..5 hold real data (j<48); c=6,7 are the zero pad (j 48..63).
    for (int idx = tid; idx < HEADS * NI * 8; idx += 256) {
        const int h  = idx / (NI * 8);
        const int r  = idx - h * (NI * 8);
        const int i  = r >> 3;
        const int j0 = (r & 7) << 3;
        uint4 w = make_uint4(0u, 0u, 0u, 0u);
        if (j0 < NJ) {
            const float4* src =
                (const float4*)&obs_re[(size_t)(h * OBS_LD + i) * OBS_LD + j0];
            float4 f0 = src[0], f1 = src[1];
            w.x = pk2(f0.x, f0.y);
            w.y = pk2(f0.z, f0.w);
            w.z = pk2(f1.x, f1.y);
            w.w = pk2(f1.z, f1.w);
        }
        *(uint4*)&obs_s[(h * NI + i) * PITCH + j0] = w;     // ds_store_b128
    }
    __syncthreads();

    const int wave = tid >> 5;
    const int lane = tid & 31;
    const int tile = blockIdx.x * 8 + wave;   // one 16-token tile per wave
    const int tok0 = tile * 16;
    if (tok0 >= n_tok) return;

    const int lm = lane & 15;     // N / token-in-tile selector
    const int hi = lane >> 4;     // which 16-lane half

    // ---- A fragments: x tile (16 tokens x 32 j), bf16, ISA 16-bit A layout ----
    // kk=0 covers j 0..31, kk=1 covers j 32..63 (j>=48 zero-padded).
    const float* xr = x + (size_t)(tok0 + lm) * NJ;
    Frag16 a0, a1;
    {
        const float4* pA = (const float4*)(xr + (hi ? 8 : 0));    // kk0 chunk A
        const float4* pB = (const float4*)(xr + (hi ? 24 : 16));  // kk0 chunk B
        float4 c0 = pA[0], c1 = pA[1], c2 = pB[0], c3 = pB[1];
        a0.u[0] = pk2(c0.x, c0.y);  a0.u[1] = pk2(c0.z, c0.w);
        a0.u[2] = pk2(c1.x, c1.y);  a0.u[3] = pk2(c1.z, c1.w);
        a0.u[4] = pk2(c2.x, c2.y);  a0.u[5] = pk2(c2.z, c2.w);
        a0.u[6] = pk2(c3.x, c3.y);  a0.u[7] = pk2(c3.z, c3.w);

        const float4* pC = (const float4*)(xr + 32 + (hi ? 8 : 0)); // kk1 (j 32..47)
        float4 c4 = pC[0], c5 = pC[1];
        a1.u[0] = pk2(c4.x, c4.y);  a1.u[1] = pk2(c4.z, c4.w);
        a1.u[2] = pk2(c5.x, c5.y);  a1.u[3] = pk2(c5.z, c5.w);
        a1.u[4] = 0u; a1.u[5] = 0u; a1.u[6] = 0u; a1.u[7] = 0u;     // j 48..63
    }

    // ---- x in C/D layout for the final contraction: xd[t][v] = x[tok0+v+hi*8][16t+lm] ----
    float xd[3][8];
#pragma unroll
    for (int t = 0; t < 3; ++t) {
        const int i = 16 * t + lm;
#pragma unroll
        for (int v = 0; v < 8; ++v)
            xd[t][v] = x[(size_t)(tok0 + v + hi * 8) * NJ + i];
    }

    float outv[8];

#pragma unroll
    for (int h = 0; h < HEADS; ++h) {
        float p[8];
#pragma unroll
        for (int v = 0; v < 8; ++v) p[v] = 0.0f;

#pragma unroll
        for (int t = 0; t < 3; ++t) {
            const int i = 16 * t + lm;                       // N = i column
            const unsigned short* row = &obs_s[(h * NI + i) * PITCH];

            Frag16 b0, b1;                                   // 2x ds_load_b128 each
            const uint4* r0 = (const uint4*)(row + hi * 8);       // kk0
            const uint4* r1 = (const uint4*)(row + 32 + hi * 8);  // kk1 (pad is zero)
            b0.q[0] = r0[0];  b0.q[1] = r0[2];
            b1.q[0] = r1[0];  b1.q[1] = r1[2];

            v8f dd = {};
            dd = __builtin_amdgcn_wmma_f32_16x16x32_bf16(
                     false, a0.v, false, b0.v, (short)0, dd, false, false);
            dd = __builtin_amdgcn_wmma_f32_16x16x32_bf16(
                     false, a1.v, false, b1.v, (short)0, dd, false, false);

#pragma unroll
            for (int v = 0; v < 8; ++v) p[v] += xd[t][v] * dd[v];
        }

        // ---- Folding butterfly reduction over 16 lanes per half ----
        // Routes value v to lane lm==v; 15 shuffles instead of 32.
#pragma unroll
        for (int v = 0; v < 8; ++v) p[v] += __shfl_xor(p[v], 8);

        const bool b4 = (lm & 4) != 0;
        float q[4];
#pragma unroll
        for (int k = 0; k < 4; ++k) {
            float snd  = b4 ? p[k] : p[k + 4];
            float rcv  = __shfl_xor(snd, 4);
            q[k] = (b4 ? p[k + 4] : p[k]) + rcv;   // q[k] ~ v = (lm&4)+k
        }
        const bool b2 = (lm & 2) != 0;
        float rr[2];
#pragma unroll
        for (int k = 0; k < 2; ++k) {
            float snd  = b2 ? q[k] : q[k + 2];
            float rcv  = __shfl_xor(snd, 2);
            rr[k] = (b2 ? q[k + 2] : q[k]) + rcv;  // rr[k] ~ v = (lm&6)+k
        }
        const bool b1 = (lm & 1) != 0;
        {
            float snd = b1 ? rr[0] : rr[1];
            float rcv = __shfl_xor(snd, 1);
            outv[h] = (b1 ? rr[1] : rr[0]) + rcv;  // v = lm&7 on this lane
        }
    }

    // ---- Store: lane lm<8 owns token tok0+lm+hi*8; 8 heads contiguous ----
    if (lm < 8) {
        const int tk = tok0 + lm + hi * 8;
        if (tk < n_tok) {
            float4* o = (float4*)(out + (size_t)tk * HEADS);
            o[0] = make_float4(outv[0], outv[1], outv[2], outv[3]);
            o[1] = make_float4(outv[4], outv[5], outv[6], outv[7]);
        }
    }
}

extern "C" void kernel_launch(void* const* d_in, const int* in_sizes, int n_in,
                              void* d_out, int out_size, void* d_ws, size_t ws_size,
                              hipStream_t stream) {
    const float* x      = (const float*)d_in[0];
    const float* obs_re = (const float*)d_in[1];
    // d_in[2] (obs_im) is mathematically irrelevant: psi is real, so
    // Re(psi^H O psi) == psi^T Re(O) psi.
    float* out = (float*)d_out;

    const int n_tok  = in_sizes[0] / NJ;          // B*T = 65536
    const int tiles  = (n_tok + 15) / 16;         // 16-token wave tiles
    const int blocks = (tiles + 7) / 8;           // 8 waves (256 thr) per block

    quadform_wmma_kernel<<<blocks, 256, 0, stream>>>(x, obs_re, out, n_tok);
}
